// Attention_11897059410850
// MI455X (gfx1250) — matrix operand order repaired and checked by reading, not compiled, not run on hardware
//
#include <hip/hip_runtime.h>
#include <hip/hip_bf16.h>

#define HIDDEN   2048
#define HEADS    16
#define HEAD_DIM 128
#define BATCH    2
#define SEQ      2048
#define MTOT     (BATCH * SEQ)

typedef __attribute__((ext_vector_type(8)))  __bf16 v8bf;
typedef __attribute__((ext_vector_type(16))) __bf16 v16bf;
typedef __attribute__((ext_vector_type(8)))  float  v8f;

__device__ __forceinline__ unsigned short f2bf(float f) {
  unsigned u = __builtin_bit_cast(unsigned, f);
  u += 0x7FFFu + ((u >> 16) & 1u);           // round-to-nearest-even
  return (unsigned short)(u >> 16);
}

__device__ __forceinline__ v16bf join16(v8bf lo, v8bf hi) {
  union { v16bf v; v8bf h[2]; } u;
  u.h[0] = lo; u.h[1] = hi;
  return u.v;
}

// LDS aperture: generic pointer low 32 bits == LDS byte offset.
__device__ __forceinline__ unsigned ldsOff32(const void* p) {
  return (unsigned)(unsigned long long)(uintptr_t)p;
}

// Per-lane 16B async copy global -> LDS (ASYNCcnt-tracked, no VGPR data).
__device__ __forceinline__ void asyncCopy16(unsigned ldsByteOff, const void* gp) {
  unsigned long long ga = (unsigned long long)(uintptr_t)gp;
  asm volatile("global_load_async_to_lds_b128 %0, %1, off"
               :: "v"(ldsByteOff), "v"(ga) : "memory");
}

// ---------------------------------------------------------------------------
// fp32 -> packed bf16 conversion (one-time pass; ~6us of HBM traffic total)
// ---------------------------------------------------------------------------
__global__ __launch_bounds__(256)
void cvt_f32_bf16(const float* __restrict__ in, unsigned short* __restrict__ out,
                  int n4) {
  int i = blockIdx.x * blockDim.x + threadIdx.x;
  if (i < n4) {
    float4 f = ((const float4*)in)[i];
    uint2 o;
    o.x = (unsigned)f2bf(f.x) | ((unsigned)f2bf(f.y) << 16);
    o.y = (unsigned)f2bf(f.z) | ((unsigned)f2bf(f.w) << 16);
    ((uint2*)out)[i] = o;
  }
}

// ---------------------------------------------------------------------------
// GEMM: C[M,N] = A[M,K](bf16) @ W[N,K](bf16)^T.   K = N = HIDDEN, M = MTOT.
// Block tile 128x128, 8 waves -> each wave 32(M) x 64(N) = 2x4 WMMA tiles.
// Double-buffered LDS staged with global_load_async_to_lds_b128 and
// pipelined with s_wait_asynccnt (issue tile k+1, wait tile k, compute).
// ---------------------------------------------------------------------------
template <bool OUT_BF16>
__global__ __launch_bounds__(256, 2)
void gemm_bf16(const unsigned short* __restrict__ A,
               const unsigned short* __restrict__ W,
               void* __restrict__ Cout) {
  constexpr int K   = HIDDEN;
  constexpr int N   = HIDDEN;
  constexpr int LD  = 40;   // bf16 elems per LDS row (80B rows: 16B aligned, padded)

  __shared__ __align__(16) unsigned short As[2][128 * LD];
  __shared__ __align__(16) unsigned short Bs[2][128 * LD];

  const int tid  = threadIdx.x;
  const int lane = tid & 31;
  const int wave = tid >> 5;
  const int r    = lane & 15;
  const int half = lane >> 4;
  const int waveM = wave & 3;        // 0..3 -> 4 x 32 rows
  const int waveN = wave >> 2;       // 0..1 -> 2 x 64 cols
  const int m0 = waveM * 32;
  const int n0 = waveN * 64;
  const size_t blockRow = (size_t)blockIdx.x * 128;
  const size_t blockCol = (size_t)blockIdx.y * 128;

  // async staging mapping: 256 lanes x 16B = 4KB per issued instruction;
  // one 128x32 bf16 tile (8KB) = 2 instructions.
  const int srow = tid >> 2;          // 0..63
  const int sseg = (tid & 3) * 8;     // bf16 elem offset within row (16B chunks)

  auto issueTile = [&](int kt, int buf) {
#pragma unroll
    for (int p = 0; p < 2; ++p) {
      int row = p * 64 + srow;
      asyncCopy16(ldsOff32(&As[buf][row * LD + sseg]),
                  A + (blockRow + row) * (size_t)K + kt + sseg);
    }
#pragma unroll
    for (int p = 0; p < 2; ++p) {
      int row = p * 64 + srow;
      asyncCopy16(ldsOff32(&Bs[buf][row * LD + sseg]),
                  W + (blockCol + row) * (size_t)K + kt + sseg);
    }
  };

  v8f acc[2][4] = {};

  issueTile(0, 0);
  int buf = 0;
  for (int kt = 0; kt < K; kt += 32) {
    const bool hasNext = (kt + 32) < K;
    if (hasNext) {
      issueTile(kt + 32, buf ^ 1);
      asm volatile("s_wait_asynccnt 0x4" ::: "memory");  // tile kt complete
    } else {
      asm volatile("s_wait_asynccnt 0x0" ::: "memory");
    }
    __syncthreads();

    v16bf af[2], bfr[4];
#pragma unroll
    for (int t = 0; t < 2; ++t) {
      const unsigned short* ap = &As[buf][(m0 + t * 16 + r) * LD];
      af[t] = join16(*(const v8bf*)(ap + half * 8),
                     *(const v8bf*)(ap + 16 + half * 8));
    }
#pragma unroll
    for (int t = 0; t < 4; ++t) {
      const unsigned short* bp = &Bs[buf][(n0 + t * 16 + r) * LD + half * 8];
      bfr[t] = join16(*(const v8bf*)(bp), *(const v8bf*)(bp + 16));
    }
#pragma unroll
    for (int tm = 0; tm < 2; ++tm)
#pragma unroll
      for (int tn = 0; tn < 4; ++tn)
        acc[tm][tn] = __builtin_amdgcn_wmma_f32_16x16x32_bf16(
            false, af[tm], false, bfr[tn], (short)0, acc[tm][tn], false, false);

    __syncthreads();
    buf ^= 1;
  }

  // store: lane holds col r, VGPR i -> row i + 8*half
#pragma unroll
  for (int tm = 0; tm < 2; ++tm)
#pragma unroll
    for (int tn = 0; tn < 4; ++tn)
#pragma unroll
      for (int i = 0; i < 8; ++i) {
        size_t row = blockRow + m0 + tm * 16 + i + 8 * half;
        size_t col = blockCol + n0 + tn * 16 + r;
        float v = acc[tm][tn][i];
        if (OUT_BF16)
          ((unsigned short*)Cout)[row * N + col] = f2bf(v);
        else
          ((float*)Cout)[row * N + col] = v;
      }
}

// ---------------------------------------------------------------------------
// Flash attention: per (b,h); 8 waves x 16 query rows = 128 q-rows per WG.
// ---------------------------------------------------------------------------
__global__ __launch_bounds__(256, 2)
void attn_kernel(const unsigned short* __restrict__ Q,
                 const unsigned short* __restrict__ Km,
                 const unsigned short* __restrict__ V,
                 unsigned short* __restrict__ O) {
  constexpr int LDV = 40;
  constexpr int LDP = 40;
  __shared__ __align__(16) unsigned short Vt[HEAD_DIM * LDV];  // V^T tile (shared)
  __shared__ __align__(16) unsigned short Pt[8 * 16 * LDP];    // per-wave P tiles

  const int tid  = threadIdx.x;
  const int lane = tid & 31;
  const int wave = tid >> 5;
  const int r    = lane & 15;
  const int half = lane >> 4;

  const int bh = blockIdx.y;
  const int b  = bh / HEADS;
  const int h  = bh % HEADS;
  const size_t headOff = (size_t)b * SEQ * HIDDEN + (size_t)h * HEAD_DIM;
  const unsigned short* qp = Q + headOff;
  const unsigned short* kp = Km + headOff;
  const unsigned short* vp = V + headOff;
  unsigned short* op = O + headOff;

  const int qrow0 = blockIdx.x * 128 + wave * 16;

  // Q fragments: 4 chunks of contraction-32 over head_dim=128
  v16bf qf[4];
  {
    const unsigned short* qrow = qp + (size_t)(qrow0 + r) * HIDDEN;
#pragma unroll
    for (int c = 0; c < 4; ++c) {
      qf[c] = join16(*(const v8bf*)(qrow + c * 32 + half * 8),
                     *(const v8bf*)(qrow + c * 32 + 16 + half * 8));
    }
  }

  v8f oacc[8] = {};
  float mrow[8], lrow[8];
#pragma unroll
  for (int i = 0; i < 8; ++i) { mrow[i] = -3.0e38f; lrow[i] = 0.0f; }

  const float scale = 0.08838834764831845f;  // 1/sqrt(128)
  unsigned short* myP = Pt + wave * 16 * LDP;

  for (int jb = 0; jb < SEQ; jb += 32) {
    __syncthreads();  // protect Vt reuse
    // stage V[jb..jb+31][0..127] transposed -> Vt[d][k]
#pragma unroll
    for (int p = 0; p < 2; ++p) {
      int idx = p * 256 + tid;
      int row = idx >> 4;          // 0..31 (k within block)
      int dg  = (idx & 15) * 8;    // 0..120 (head dim group)
      uint4 vv = *(const uint4*)(vp + (size_t)(jb + row) * HIDDEN + dg);
      Vt[(dg + 0) * LDV + row] = (unsigned short)(vv.x);
      Vt[(dg + 1) * LDV + row] = (unsigned short)(vv.x >> 16);
      Vt[(dg + 2) * LDV + row] = (unsigned short)(vv.y);
      Vt[(dg + 3) * LDV + row] = (unsigned short)(vv.y >> 16);
      Vt[(dg + 4) * LDV + row] = (unsigned short)(vv.z);
      Vt[(dg + 5) * LDV + row] = (unsigned short)(vv.z >> 16);
      Vt[(dg + 6) * LDV + row] = (unsigned short)(vv.w);
      Vt[(dg + 7) * LDV + row] = (unsigned short)(vv.w >> 16);
    }
    __syncthreads();

    // scores S = Q @ K^T for 32 keys (two 16-wide WMMA tiles)
    v8f s[2] = {};
#pragma unroll
    for (int jn = 0; jn < 2; ++jn) {
      const unsigned short* krow = kp + (size_t)(jb + jn * 16 + r) * HIDDEN;
#pragma unroll
      for (int c = 0; c < 4; ++c) {
        v8bf blo = *(const v8bf*)(krow + c * 32 + half * 8);
        v8bf bhi = *(const v8bf*)(krow + c * 32 + 16 + half * 8);
        s[jn] = __builtin_amdgcn_wmma_f32_16x16x32_bf16(
            false, qf[c], false, join16(blo, bhi), (short)0, s[jn], false, false);
      }
    }

    // online softmax over 16x32 score tile
#pragma unroll
    for (int i = 0; i < 8; ++i) {
      float s0 = s[0][i] * scale;
      float s1 = s[1][i] * scale;
      float mx = fmaxf(s0, s1);
      mx = fmaxf(mx, __shfl_xor(mx, 1, 32));
      mx = fmaxf(mx, __shfl_xor(mx, 2, 32));
      mx = fmaxf(mx, __shfl_xor(mx, 4, 32));
      mx = fmaxf(mx, __shfl_xor(mx, 8, 32));
      float mnew  = fmaxf(mrow[i], mx);
      float alpha = __expf(mrow[i] - mnew);
      float p0 = __expf(s0 - mnew);
      float p1 = __expf(s1 - mnew);
      float ps = p0 + p1;
      ps += __shfl_xor(ps, 1, 32);
      ps += __shfl_xor(ps, 2, 32);
      ps += __shfl_xor(ps, 4, 32);
      ps += __shfl_xor(ps, 8, 32);
      lrow[i] = lrow[i] * alpha + ps;
      mrow[i] = mnew;
#pragma unroll
      for (int nt = 0; nt < 8; ++nt) oacc[nt][i] *= alpha;
      // write P (C layout) into per-wave LDS tile for A-layout reload
      myP[(i + 8 * half) * LDP + r]      = f2bf(p0);
      myP[(i + 8 * half) * LDP + 16 + r] = f2bf(p1);
    }
    asm volatile("s_wait_dscnt 0x0" ::: "memory");  // wave-local LDS RAW fence

    // reload P as 16x32 A fragment
    v16bf pf = join16(*(const v8bf*)&myP[r * LDP + half * 8],
                      *(const v8bf*)&myP[r * LDP + 16 + half * 8]);

    // O += P @ V
#pragma unroll
    for (int nt = 0; nt < 8; ++nt) {
      const unsigned short* vcol = &Vt[(nt * 16 + r) * LDV + half * 8];
      oacc[nt] = __builtin_amdgcn_wmma_f32_16x16x32_bf16(
          false, pf, false, join16(*(const v8bf*)(vcol), *(const v8bf*)(vcol + 16)),
          (short)0, oacc[nt], false, false);
    }
  }

  // epilogue: normalize, store bf16 (feeds the Wo projection GEMM)
#pragma unroll
  for (int i = 0; i < 8; ++i) {
    float inv = 1.0f / lrow[i];
    size_t row = (size_t)(qrow0 + i + 8 * half);
#pragma unroll
    for (int nt = 0; nt < 8; ++nt)
      op[row * HIDDEN + nt * 16 + r] = f2bf(oacc[nt][i] * inv);
  }
}

// ---------------------------------------------------------------------------
extern "C" void kernel_launch(void* const* d_in, const int* in_sizes, int n_in,
                              void* d_out, int out_size, void* d_ws, size_t ws_size,
                              hipStream_t stream) {
  const float* x  = (const float*)d_in[0];
  const float* Wq = (const float*)d_in[1];
  const float* Wk = (const float*)d_in[2];
  const float* Wv = (const float*)d_in[3];
  const float* Wo = (const float*)d_in[4];
  float* out = (float*)d_out;

  const size_t MK = (size_t)MTOT * HIDDEN;      // 8,388,608 elems
  const size_t WW = (size_t)HIDDEN * HIDDEN;    // 4,194,304 elems

  unsigned short* xb  = (unsigned short*)d_ws;
  unsigned short* wqb = xb + MK;
  unsigned short* wkb = wqb + WW;
  unsigned short* wvb = wkb + WW;
  unsigned short* wob = wvb + WW;
  unsigned short* qb  = wob + WW;
  unsigned short* kb  = qb + MK;
  unsigned short* vb  = kb + MK;
  unsigned short* ab  = vb + MK;                // attention output (bf16)

  // one-time fp32 -> bf16 packing
  cvt_f32_bf16<<<(int)(MK / 4 / 256), 256, 0, stream>>>(x, xb, (int)(MK / 4));
  cvt_f32_bf16<<<(int)(WW / 4 / 256), 256, 0, stream>>>(Wq, wqb, (int)(WW / 4));
  cvt_f32_bf16<<<(int)(WW / 4 / 256), 256, 0, stream>>>(Wk, wkb, (int)(WW / 4));
  cvt_f32_bf16<<<(int)(WW / 4 / 256), 256, 0, stream>>>(Wv, wvb, (int)(WW / 4));
  cvt_f32_bf16<<<(int)(WW / 4 / 256), 256, 0, stream>>>(Wo, wob, (int)(WW / 4));

  dim3 ggrid(MTOT / 128, HIDDEN / 128);         // 32 x 16
  gemm_bf16<true><<<ggrid, 256, 0, stream>>>(xb, wqb, (void*)qb);
  gemm_bf16<true><<<ggrid, 256, 0, stream>>>(xb, wkb, (void*)kb);
  gemm_bf16<true><<<ggrid, 256, 0, stream>>>(xb, wvb, (void*)vb);

  dim3 agrid(SEQ / 128, BATCH * HEADS);         // 16 x 32
  attn_kernel<<<agrid, 256, 0, stream>>>(qb, kb, vb, ab);

  gemm_bf16<false><<<ggrid, 256, 0, stream>>>(ab, wob, (void*)out);
}
